// SensorRelationGNN_51470888075948
// MI455X (gfx1250) — matrix operand order repaired and checked by reading, not compiled, not run on hardware
//
#include <hip/hip_runtime.h>
#include <hip/hip_bf16.h>
#include <math.h>

// ---------------------------------------------------------------------------
// Sizes (fixed by the reference)
// ---------------------------------------------------------------------------
#define BATCH   256
#define NSENS   32
#define WIN     256
#define HID     256
#define HEADS   4
#define CH      64          // HID / HEADS
#define EMB     64
#define TOPK    8
#define ROWS    (BATCH * NSENS)   // 8192
#define NEG_SLOPE 0.2f
#define LN_EPS    1e-5f

typedef __attribute__((ext_vector_type(16))) _Float16 v16h;
typedef __attribute__((ext_vector_type(8)))  _Float16 v8h;
typedef __attribute__((ext_vector_type(8)))  float    v8f;

// ---------------------------------------------------------------------------
// Kernel 1: dynamic graph mask (cosine sim -> top-8 -> symmetrize -> |eye)
// one block of 32 threads (one wave32), thread i owns row i
// ---------------------------------------------------------------------------
__global__ void mask_kernel(const float* __restrict__ emb,
                            int* __restrict__ mask_ws,
                            int* __restrict__ mask_out)
{
    __shared__ float semb[NSENS * EMB];
    __shared__ float nrm[NSENS];
    __shared__ float ssp[NSENS * NSENS];
    const int i = threadIdx.x;           // 0..31

    #pragma unroll 4
    for (int e = 0; e < EMB; ++e) semb[i * EMB + e] = emb[i * EMB + e];
    __syncthreads();

    float s = 0.f;
    #pragma unroll 4
    for (int e = 0; e < EMB; ++e) { float v = semb[i * EMB + e]; s += v * v; }
    nrm[i] = fmaxf(sqrtf(s), 1e-12f);
    __syncthreads();

    float row[NSENS];
    for (int j = 0; j < NSENS; ++j) {
        float d = 0.f;
        #pragma unroll 4
        for (int e = 0; e < EMB; ++e) d += semb[i * EMB + e] * semb[j * EMB + e];
        row[j] = (j == i) ? -1e9f : d / (nrm[i] * nrm[j]);
    }
    // top-8 selection (first-max like lax.top_k)
    float sp[NSENS];
    #pragma unroll
    for (int j = 0; j < NSENS; ++j) sp[j] = 0.f;
    for (int sel = 0; sel < TOPK; ++sel) {
        int   bj = 0;
        float bv = row[0];
        #pragma unroll
        for (int j = 1; j < NSENS; ++j) if (row[j] > bv) { bv = row[j]; bj = j; }
        sp[bj]  = bv;
        row[bj] = -2e9f;
    }
    #pragma unroll
    for (int j = 0; j < NSENS; ++j) ssp[i * NSENS + j] = sp[j];
    __syncthreads();

    for (int j = 0; j < NSENS; ++j) {
        float sym = 0.5f * (ssp[i * NSENS + j] + ssp[j * NSENS + i]);
        int mv = (sym != 0.f) || (i == j);
        mask_ws [i * NSENS + j] = mv;
        mask_out[i * NSENS + j] = mv;
    }
}

// ---------------------------------------------------------------------------
// fp32 -> fp16 row-major copy
// ---------------------------------------------------------------------------
__global__ void cvt_f16_kernel(const float* __restrict__ src,
                               _Float16* __restrict__ dst, int n)
{
    int idx = blockIdx.x * blockDim.x + threadIdx.x;
    if (idx < n) dst[idx] = (_Float16)src[idx];
}

// fp32 [K,N] -> fp16 transposed [N,K]  (so WMMA B lanes read contiguous K runs)
__global__ void cvt_t_f16_kernel(const float* __restrict__ src,
                                 _Float16* __restrict__ dst, int K, int N)
{
    int idx = blockIdx.x * blockDim.x + threadIdx.x;
    if (idx < K * N) {
        int n = idx / K, k = idx - n * K;
        dst[idx] = (_Float16)src[k * N + n];
    }
}

// ---------------------------------------------------------------------------
// Register-blocked wave-tile WMMA GEMM:
//   D[M,N] = act(A[M,K](f16) * Bt[N,K](f16)^T + bias)
// Each wave computes a 32(M) x 64(N) tile = 2x4 grid of 16x16 WMMA tiles.
// Per K-step of 32: 2 A-frags + 4 B-frags feed 8 v_wmma_f32_16x16x32_f16
// (1.5 b128 loads per WMMA instead of 4 -> 3x arithmetic intensity).
// A-frag layout (ISA 7.12.2, 16-bit A 16x32): lane = m + 16*g,
//   halves 0..7  <-> k = g*8 + 0..7 ; halves 8..15 <-> k = 16 + g*8 + 0..7
// B-frag layout (mirrors A): lane = n + 16*g, halves 0..7 <-> k = g*8 + 0..7 ; halves 8..15 <-> k = 16 + g*8 + 0..7
// C/D layout: vgpr r, lane L -> (m = r + 8*(L/16), n = L%16)
// Requires: M % 32 == 0, N % 64 == 0, K % 32 == 0 (true for all uses here).
// ---------------------------------------------------------------------------
#define TM 2
#define TN 4

__global__ void wmma_gemm_kernel(const _Float16* __restrict__ A, int lda,
                                 const _Float16* __restrict__ Bt, int ldb,
                                 const float* __restrict__ bias,
                                 float* __restrict__ outF32,
                                 _Float16* __restrict__ outF16,
                                 int M, int N, int K, int act)
{
    const int lane  = threadIdx.x & 31;
    const int wave  = threadIdx.x >> 5;
    const int tile  = blockIdx.x * (blockDim.x >> 5) + wave;
    const int tilesN = N >> 6;                  // 64-wide column tiles
    const int tm = tile / tilesN;
    const int tn = tile - tm * tilesN;
    if (tm * 32 >= M) return;

    const int mloc = lane & 15;
    const int g    = lane >> 4;

    const _Float16* arow[TM];
    const _Float16* brow[TN];
    #pragma unroll
    for (int mt = 0; mt < TM; ++mt)
        arow[mt] = A  + (size_t)(tm * 32 + mt * 16 + mloc) * lda;
    #pragma unroll
    for (int nt = 0; nt < TN; ++nt)
        brow[nt] = Bt + (size_t)(tn * 64 + nt * 16 + mloc) * ldb;

    v8f acc[TM][TN];
    #pragma unroll
    for (int mt = 0; mt < TM; ++mt)
        #pragma unroll
        for (int nt = 0; nt < TN; ++nt)
            acc[mt][nt] = (v8f){};

    for (int k0 = 0; k0 < K; k0 += 32) {
        v16h a[TM], b[TN];
        #pragma unroll
        for (int mt = 0; mt < TM; ++mt) {
            v8h lo = *(const v8h*)(arow[mt] + k0 + g * 8);
            v8h hi = *(const v8h*)(arow[mt] + k0 + 16 + g * 8);
            #pragma unroll
            for (int u = 0; u < 8; ++u) { a[mt][u] = lo[u]; a[mt][u + 8] = hi[u]; }
        }
        #pragma unroll
        for (int nt = 0; nt < TN; ++nt) {
            v8h lo = *(const v8h*)(brow[nt] + k0 + g * 8);
            v8h hi = *(const v8h*)(brow[nt] + k0 + 16 + g * 8);
            #pragma unroll
            for (int u = 0; u < 8; ++u) { b[nt][u] = lo[u]; b[nt][u + 8] = hi[u]; }
        }
        #pragma unroll
        for (int mt = 0; mt < TM; ++mt)
            #pragma unroll
            for (int nt = 0; nt < TN; ++nt)
                acc[mt][nt] = __builtin_amdgcn_wmma_f32_16x16x32_f16(
                    /*neg_a=*/false, a[mt], /*neg_b=*/false, b[nt],
                    /*c_mod=*/(short)0, acc[mt][nt],
                    /*reuse_a=*/false, /*reuse_b=*/false);
    }

    // epilogue: bias + activation + store (f32 and/or f16)
    #pragma unroll
    for (int nt = 0; nt < TN; ++nt) {
        const int n  = tn * 64 + nt * 16 + (lane & 15);
        const float bv = bias ? bias[n] : 0.f;
        #pragma unroll
        for (int mt = 0; mt < TM; ++mt) {
            #pragma unroll
            for (int r = 0; r < 8; ++r) {
                const int m = tm * 32 + mt * 16 + r + 8 * g;
                float v = acc[mt][nt][r] + bv;
                if (act == 1) v = v > 0.f ? v : 0.f;          // ReLU
                if (outF32) outF32[(size_t)m * N + n] = v;
                if (outF16) outF16[(size_t)m * N + n] = (_Float16)v;
            }
        }
    }
}

// ---------------------------------------------------------------------------
// GATv2 layer 1 attention (heads=4, C=64) + bias + ELU, emit f16
// block = 128 threads = 4 waves; wave h owns head h, lane = source node j
// ---------------------------------------------------------------------------
__global__ void att1_kernel(const float* __restrict__ xl,
                            const float* __restrict__ xr,
                            const float* __restrict__ att,
                            const float* __restrict__ bias,
                            const int* __restrict__ mask,
                            _Float16* __restrict__ h1)
{
    __shared__ float s_att[HID];
    __shared__ float s_alpha[HEADS][NSENS];
    const int bi = blockIdx.x;          // b*32 + i
    const int b  = bi >> 5, i = bi & 31;
    const int t  = threadIdx.x;         // 0..127
    s_att[t]        = att[t];
    s_att[t + 128]  = att[t + 128];
    __syncthreads();

    const int h = t >> 5, j = t & 31;
    const float* xr_i = xr + (size_t)bi * HID + h * CH;
    const float* xl_j = xl + (size_t)(b * NSENS + j) * HID + h * CH;

    float logit = 0.f;
    #pragma unroll 4
    for (int c = 0; c < CH; ++c) {
        float s = xr_i[c] + xl_j[c];
        s = s > 0.f ? s : NEG_SLOPE * s;
        logit += s_att[h * CH + c] * s;
    }
    if (!mask[i * NSENS + j]) logit = -1e9f;

    float mx = logit;
    #pragma unroll
    for (int o = 16; o; o >>= 1) mx = fmaxf(mx, __shfl_xor(mx, o, 32));
    float e  = __expf(logit - mx);
    float se = e;
    #pragma unroll
    for (int o = 16; o; o >>= 1) se += __shfl_xor(se, o, 32);
    s_alpha[h][j] = e / se;
    __syncthreads();

    // 256 output channels, 128 threads -> 2 each
    #pragma unroll
    for (int q = 0; q < 2; ++q) {
        const int ch = t + q * 128;
        const int ho = ch >> 6, c = ch & 63;
        const float* xlb = xl + (size_t)(b * NSENS) * HID + ho * CH + c;
        float o = 0.f;
        #pragma unroll 4
        for (int j2 = 0; j2 < NSENS; ++j2)
            o += s_alpha[ho][j2] * xlb[(size_t)j2 * HID];
        o += bias[ch];
        o = o > 0.f ? o : (__expf(o) - 1.0f);          // ELU (after bias)
        h1[(size_t)bi * HID + ch] = (_Float16)o;
    }
}

// ---------------------------------------------------------------------------
// GATv2 layer 2 attention (heads=1, C=256) + bias + LayerNorm, emit f16
// block = 32 threads (one wave32); lane = source node j, then lane owns 8 ch
// ---------------------------------------------------------------------------
__global__ void att2_ln_kernel(const float* __restrict__ xl,
                               const float* __restrict__ xr,
                               const float* __restrict__ att2,
                               const float* __restrict__ bias2,
                               const float* __restrict__ gamma,
                               const float* __restrict__ beta,
                               const int* __restrict__ mask,
                               _Float16* __restrict__ hn)
{
    __shared__ float s_alpha[NSENS];
    const int bi = blockIdx.x;
    const int b  = bi >> 5, i = bi & 31;
    const int j  = threadIdx.x;          // 0..31

    const float* xr_i = xr + (size_t)bi * HID;
    const float* xl_j = xl + (size_t)(b * NSENS + j) * HID;

    float logit = 0.f;
    #pragma unroll 4
    for (int c = 0; c < HID; ++c) {
        float s = xr_i[c] + xl_j[c];
        s = s > 0.f ? s : NEG_SLOPE * s;
        logit += att2[c] * s;
    }
    if (!mask[i * NSENS + j]) logit = -1e9f;

    float mx = logit;
    #pragma unroll
    for (int o = 16; o; o >>= 1) mx = fmaxf(mx, __shfl_xor(mx, o, 32));
    float e  = __expf(logit - mx);
    float se = e;
    #pragma unroll
    for (int o = 16; o; o >>= 1) se += __shfl_xor(se, o, 32);
    s_alpha[j] = e / se;
    __syncthreads();

    float v[8], sum = 0.f, sumsq = 0.f;
    #pragma unroll
    for (int q = 0; q < 8; ++q) {
        const int c = j * 8 + q;
        float o = 0.f;
        #pragma unroll 4
        for (int j2 = 0; j2 < NSENS; ++j2)
            o += s_alpha[j2] * xl[(size_t)(b * NSENS + j2) * HID + c];
        o += bias2[c];
        v[q] = o; sum += o; sumsq += o * o;
    }
    #pragma unroll
    for (int o = 16; o; o >>= 1) {
        sum   += __shfl_xor(sum,   o, 32);
        sumsq += __shfl_xor(sumsq, o, 32);
    }
    const float mu  = sum * (1.f / HID);
    const float var = sumsq * (1.f / HID) - mu * mu;
    const float rs  = rsqrtf(var + LN_EPS);
    #pragma unroll
    for (int q = 0; q < 8; ++q) {
        const int c = j * 8 + q;
        hn[(size_t)bi * HID + c] = (_Float16)((v[q] - mu) * rs * gamma[c] + beta[c]);
    }
}

// ---------------------------------------------------------------------------
// combined = concat([hnorm, x], axis=-1) in f16
// ---------------------------------------------------------------------------
__global__ void concat_kernel(const _Float16* __restrict__ hn,
                              const _Float16* __restrict__ x16,
                              _Float16* __restrict__ comb)
{
    const size_t idx = (size_t)blockIdx.x * blockDim.x + threadIdx.x;
    if (idx < (size_t)ROWS * (HID + WIN)) {
        const size_t r = idx >> 9;          // / 512
        const int    c = (int)(idx & 511);
        comb[idx] = (c < HID) ? hn[r * HID + c] : x16[r * WIN + (c - HID)];
    }
}

// ---------------------------------------------------------------------------
// pred = hid @ w2 + b2  (N=1 -> wave-per-row dot)
// ---------------------------------------------------------------------------
__global__ void head2_kernel(const _Float16* __restrict__ hid,
                             const float* __restrict__ w2,
                             const float* __restrict__ b2,
                             float* __restrict__ pred)
{
    const int row  = blockIdx.x * (blockDim.x >> 5) + (threadIdx.x >> 5);
    const int lane = threadIdx.x & 31;
    if (row >= ROWS) return;
    float s = 0.f;
    #pragma unroll
    for (int c = lane; c < HID; c += 32)
        s += (float)hid[(size_t)row * HID + c] * w2[c];
    #pragma unroll
    for (int o = 16; o; o >>= 1) s += __shfl_xor(s, o, 32);
    if (lane == 0) pred[row] = s + b2[0];
}

// ---------------------------------------------------------------------------
// host-side launch
// ---------------------------------------------------------------------------
extern "C" void kernel_launch(void* const* d_in, const int* in_sizes, int n_in,
                              void* d_out, int out_size, void* d_ws, size_t ws_size,
                              hipStream_t stream)
{
    const float* x        = (const float*)d_in[0];
    const float* node_emb = (const float*)d_in[1];
    const float* tw   = (const float*)d_in[2];
    const float* tb   = (const float*)d_in[3];
    const float* wl1  = (const float*)d_in[4];
    const float* bl1  = (const float*)d_in[5];
    const float* wr1  = (const float*)d_in[6];
    const float* br1  = (const float*)d_in[7];
    const float* att1 = (const float*)d_in[8];
    const float* bias1= (const float*)d_in[9];
    const float* wl2  = (const float*)d_in[10];
    const float* bl2  = (const float*)d_in[11];
    const float* wr2  = (const float*)d_in[12];
    const float* br2  = (const float*)d_in[13];
    const float* att2 = (const float*)d_in[14];
    const float* bias2= (const float*)d_in[15];
    const float* gamma= (const float*)d_in[16];
    const float* beta = (const float*)d_in[17];
    const float* w1   = (const float*)d_in[18];
    const float* b1   = (const float*)d_in[19];
    const float* w2   = (const float*)d_in[20];
    const float* b2   = (const float*)d_in[21];

    float* pred_out = (float*)d_out;                 // ROWS floats
    int*   mask_out = (int*)(pred_out + ROWS);       // 32*32 int32

    // ------- workspace layout (bytes) -------
    char* ws = (char*)d_ws;
    size_t off = 0;
    auto alloc = [&](size_t bytes) { char* p = ws + off; off += (bytes + 255) & ~(size_t)255; return p; };
    _Float16* x16     = (_Float16*)alloc((size_t)ROWS * WIN * 2);        // 4 MB
    _Float16* feat16  = (_Float16*)alloc((size_t)ROWS * HID * 2);        // 4 MB
    _Float16* h1_16   = (_Float16*)alloc((size_t)ROWS * HID * 2);        // 4 MB
    _Float16* hn16    = (_Float16*)alloc((size_t)ROWS * HID * 2);        // 4 MB
    _Float16* hid16   = (_Float16*)alloc((size_t)ROWS * HID * 2);        // 4 MB
    _Float16* comb16  = (_Float16*)alloc((size_t)ROWS * (HID + WIN) * 2);// 8 MB
    float*    xlf     = (float*)   alloc((size_t)ROWS * HID * 4);        // 8 MB
    float*    xrf     = (float*)   alloc((size_t)ROWS * HID * 4);        // 8 MB
    _Float16* tw_t    = (_Float16*)alloc((size_t)WIN * HID * 2);
    _Float16* wl1_t   = (_Float16*)alloc((size_t)HID * HID * 2);
    _Float16* wr1_t   = (_Float16*)alloc((size_t)HID * HID * 2);
    _Float16* wl2_t   = (_Float16*)alloc((size_t)HID * HID * 2);
    _Float16* wr2_t   = (_Float16*)alloc((size_t)HID * HID * 2);
    _Float16* w1_t    = (_Float16*)alloc((size_t)(HID + WIN) * HID * 2);
    int*      mask_ws = (int*)     alloc((size_t)NSENS * NSENS * 4);
    (void)ws_size; (void)n_in; (void)in_sizes; (void)out_size;

    // 1) graph mask
    mask_kernel<<<1, 32, 0, stream>>>(node_emb, mask_ws, mask_out);

    // 2) conversions
    {
        int n = ROWS * WIN;
        cvt_f16_kernel<<<(n + 255) / 256, 256, 0, stream>>>(x, x16, n);
        int kn = WIN * HID;
        cvt_t_f16_kernel<<<(kn + 255) / 256, 256, 0, stream>>>(tw,  tw_t,  WIN, HID);
        kn = HID * HID;
        cvt_t_f16_kernel<<<(kn + 255) / 256, 256, 0, stream>>>(wl1, wl1_t, HID, HID);
        cvt_t_f16_kernel<<<(kn + 255) / 256, 256, 0, stream>>>(wr1, wr1_t, HID, HID);
        cvt_t_f16_kernel<<<(kn + 255) / 256, 256, 0, stream>>>(wl2, wl2_t, HID, HID);
        cvt_t_f16_kernel<<<(kn + 255) / 256, 256, 0, stream>>>(wr2, wr2_t, HID, HID);
        kn = (HID + WIN) * HID;
        cvt_t_f16_kernel<<<(kn + 255) / 256, 256, 0, stream>>>(w1, w1_t, HID + WIN, HID);
    }

    // GEMM launch helper: wave tiles of 32x64 -> tiles = (M/32)*(N/64),
    // 4 wave-tiles per 128-thread block
    auto gemm = [&](const _Float16* A, int lda, const _Float16* Bt, int ldb,
                    const float* bias, float* oF32, _Float16* oF16,
                    int M, int N, int K, int act) {
        int tiles  = (M / 32) * (N / 64);
        int blocks = (tiles + 3) / 4;
        wmma_gemm_kernel<<<blocks, 128, 0, stream>>>(A, lda, Bt, ldb, bias,
                                                     oF32, oF16, M, N, K, act);
    };

    // 3) feat = relu(x @ tw + tb)   -> f16
    gemm(x16, WIN, tw_t, WIN, tb, nullptr, feat16, ROWS, HID, WIN, 1);

    // 4) layer-1 projections
    gemm(feat16, HID, wl1_t, HID, bl1, xlf, nullptr, ROWS, HID, HID, 0);
    gemm(feat16, HID, wr1_t, HID, br1, xrf, nullptr, ROWS, HID, HID, 0);

    // 5) layer-1 attention + ELU -> h1 (f16)
    att1_kernel<<<ROWS, 128, 0, stream>>>(xlf, xrf, att1, bias1, mask_ws, h1_16);

    // 6) layer-2 projections (reuse xlf/xrf)
    gemm(h1_16, HID, wl2_t, HID, bl2, xlf, nullptr, ROWS, HID, HID, 0);
    gemm(h1_16, HID, wr2_t, HID, br2, xrf, nullptr, ROWS, HID, HID, 0);

    // 7) layer-2 attention + bias + LayerNorm -> hn (f16)
    att2_ln_kernel<<<ROWS, 32, 0, stream>>>(xlf, xrf, att2, bias2, gamma, beta,
                                            mask_ws, hn16);

    // 8) combined = [hn, x]
    {
        size_t n = (size_t)ROWS * (HID + WIN);
        concat_kernel<<<(unsigned)((n + 255) / 256), 256, 0, stream>>>(hn16, x16, comb16);
    }

    // 9) hid = relu(combined @ w1 + b1) -> f16
    gemm(comb16, HID + WIN, w1_t, HID + WIN, b1, nullptr, hid16, ROWS, HID, HID + WIN, 1);

    // 10) pred = hid @ w2 + b2
    head2_kernel<<<ROWS / 8, 256, 0, stream>>>(hid16, w2, b2, pred_out);
}
